// MultiHeadAttention_44925357916764
// MI455X (gfx1250) — matrix operand order repaired and checked by reading, not compiled, not run on hardware
//
#include <hip/hip_runtime.h>
#include <hip/hip_bf16.h>
#include <type_traits>

// MHA: B=4, S=2048, D_MODEL=1024, H=16, DEPTH=64
// d_out = [out (8192*1024 f32)] ++ [attn (4*16*2048*2048 f32)]
// ws: qh bf16 [B,H,S,64] @0 | kh bf16 @16MB | v^T bf16 [B,H,64,S] @32MB | concat bf16 @48MB

#define MASK_NEG -1.0e9f

typedef __attribute__((ext_vector_type(16))) __bf16 v16bf;
typedef __attribute__((ext_vector_type(8)))  float  v8f;

struct B32Frag { uint4 lo, hi; };   // 32 bytes = 16 bf16

static __device__ inline v16bf frag_cast(uint4 lo, uint4 hi) {
  B32Frag t{lo, hi};
  return __builtin_bit_cast(v16bf, t);
}

static __device__ inline unsigned int pkbf(float a, float b) {
  unsigned int ua = __builtin_bit_cast(unsigned int, a);
  unsigned int ub = __builtin_bit_cast(unsigned int, b);
  ua = (ua + 0x7FFFu + ((ua >> 16) & 1u)) >> 16;          // RNE f32->bf16
  ub = (ub + 0x7FFFu + ((ub >> 16) & 1u)) >> 16;
  return ua | (ub << 16);
}
static __device__ inline unsigned short f2bf(float a) {
  unsigned int ua = __builtin_bit_cast(unsigned int, a);
  return (unsigned short)((ua + 0x7FFFu + ((ua >> 16) & 1u)) >> 16);
}

// Branch-free erf (Abramowitz-Stegun 7.1.26, |err| <= 1.5e-7): no EXEC divergence.
static __device__ inline float erf_bfree(float x) {
  const float ax = fabsf(x);
  const float t  = __builtin_amdgcn_rcpf(1.0f + 0.3275911f * ax);
  float y = 1.061405429f;
  y = y * t - 1.453152027f;
  y = y * t + 1.421413741f;
  y = y * t - 0.284496736f;
  y = y * t + 0.254829592f;
  y = y * t;
  const float r = 1.0f - y * __expf(-ax * ax);
  unsigned int ur = __builtin_bit_cast(unsigned int, r) & 0x7FFFFFFFu;
  unsigned int sx = __builtin_bit_cast(unsigned int, x) & 0x80000000u;
  return __builtin_bit_cast(float, ur | sx);
}
static __device__ inline float gelu_exact(float x) {
  return 0.5f * x * (1.0f + erf_bfree(x * 0.70710678118654752f));
}

// ---- CDNA5 async LDS staging helpers ----
static __device__ inline void async_copy_b128(unsigned lds_off, unsigned long long gaddr) {
  asm volatile("global_load_async_to_lds_b128 %0, %1, off"
               :: "v"(lds_off), "v"(gaddr) : "memory");
}
static __device__ inline void wait_async_le4() {
  asm volatile("s_wait_asynccnt 0x4" ::: "memory");
}
static __device__ inline void wait_async_0() {
  asm volatile("s_wait_asynccnt 0x0" ::: "memory");
}

// C = A(8192x1024) * W(1024x1024) + bias, bf16 WMMA, f32 accum, SW-pipelined.
// MODE 0: bf16 out, split-heads [B,H,S,64]    (Q,K projections)
// MODE 1: bf16 out, split-heads-T [B,H,64,S]  (V projection)
// MODE 2: f32 out, row-major [8192,1024]      (output projection)
template <typename AT, int MODE>
__global__ __launch_bounds__(256) void gemm_proj(
    const AT* __restrict__ A, const float* __restrict__ W,
    const float* __restrict__ bias, void* __restrict__ outp)
{
  constexpr bool A_F32 = std::is_same<AT, float>::value;
  __shared__ alignas(16) unsigned short lds_a[64 * 32];   // [m][k] bf16
  __shared__ alignas(16) unsigned short lds_b[128 * 32];  // [n][k] bf16 (transposed)

  const int tid = threadIdx.x;
  const int lane = tid & 31, wid = tid >> 5;
  const int waveM = wid >> 1, waveN = wid & 1;            // 4x2 wave grid
  const int lane15 = lane & 15, kg = lane >> 4;
  const int bm = blockIdx.x, bn = blockIdx.y;             // 128 x 8 blocks

  const v8f vzero = {0.f,0.f,0.f,0.f,0.f,0.f,0.f,0.f};
  v8f acc[4];
#pragma unroll
  for (int t = 0; t < 4; ++t) acc[t] = vzero;

  const int arow = tid >> 2;            // 0..63
  const int acol = (tid & 3) * 8;       // 0,8,16,24
  const int bk   = tid & 31;            // 0..31
  const int bn0  = (tid >> 5) * 16;     // 0..112

  // prefetch registers
  float4 af0{}, af1{}; uint4 au{};
  float4 w0{}, w1{}, w2{}, w3{};

  auto loadA = [&](int kk) {
    size_t off = (size_t)(bm * 64 + arow) * 1024 + kk + acol;
    if constexpr (A_F32) {
      const float4* p = (const float4*)(A + off);
      af0 = p[0]; af1 = p[1];
    } else {
      au = *(const uint4*)(A + off);
    }
  };
  auto storeA = [&]() {
    if constexpr (A_F32) {
      uint4 o;
      o.x = pkbf(af0.x, af0.y); o.y = pkbf(af0.z, af0.w);
      o.z = pkbf(af1.x, af1.y); o.w = pkbf(af1.z, af1.w);
      *(uint4*)&lds_a[arow * 32 + acol] = o;
    } else {
      *(uint4*)&lds_a[arow * 32 + acol] = au;
    }
  };
  auto loadW = [&](int kk) {
    const float4* p = (const float4*)(W + (size_t)(kk + bk) * 1024 + bn * 128 + bn0);
    w0 = p[0]; w1 = p[1]; w2 = p[2]; w3 = p[3];
  };
  auto storeW = [&]() {
    float wv[16];
    *(float4*)&wv[0]  = w0; *(float4*)&wv[4]  = w1;
    *(float4*)&wv[8]  = w2; *(float4*)&wv[12] = w3;
#pragma unroll
    for (int i = 0; i < 16; ++i) lds_b[(bn0 + i) * 32 + bk] = f2bf(wv[i]);
  };

  // prologue
  loadA(0); loadW(0);
  storeA(); storeW();
  __syncthreads();

  for (int kk = 0; kk < 1024; kk += 32) {
    const bool has_next = (kk + 32) < 1024;
    if (has_next) { loadA(kk + 32); loadW(kk + 32); }   // in flight during WMMA

    const int ra = waveM * 16 + lane15;
    v16bf va = frag_cast(*(const uint4*)&lds_a[ra * 32 + 8 * kg],
                         *(const uint4*)&lds_a[ra * 32 + 16 + 8 * kg]);
#pragma unroll
    for (int t = 0; t < 4; ++t) {
      const int nb = (waveN * 64 + t * 16 + lane15) * 32 + 8 * kg;
      v16bf vb = frag_cast(*(const uint4*)&lds_b[nb], *(const uint4*)&lds_b[nb + 16]);
      acc[t] = __builtin_amdgcn_wmma_f32_16x16x32_bf16(false, va, false, vb,
                                                       (short)0, acc[t], false, false);
    }
    __syncthreads();
    if (has_next) {
      storeA(); storeW();
      __syncthreads();
    }
  }

  // ---- epilogue: bias + layout-specific store (C: M=r+8kg, N=lane15) ----
#pragma unroll
  for (int t = 0; t < 4; ++t) {
    const int gcol = bn * 128 + waveN * 64 + t * 16 + lane15;
    const float bv = bias[gcol];
#pragma unroll
    for (int r = 0; r < 8; ++r) {
      const int grow = bm * 64 + waveM * 16 + 8 * kg + r;
      float val = acc[t][r] + bv;
      if constexpr (MODE == 2) {
        ((float*)outp)[(size_t)grow * 1024 + gcol] = val;
      } else {
        const int b = grow >> 11, s = grow & 2047;
        const int h = gcol >> 6,  d = gcol & 63;
        size_t off;
        if constexpr (MODE == 0) off = (((size_t)(b * 16 + h)) * 2048 + s) * 64 + d;
        else                     off = (((size_t)(b * 16 + h)) * 64 + d) * 2048 + s;
        ((unsigned short*)outp)[off] = f2bf(val);
      }
    }
  }
}

// Fused attention: logits = gelu(QK^T/8) + causal(-1e9); attn = softmax; ctx = attn*V.
// Block = 128 threads (4 waves), each wave owns 16 query rows; grid = (32 qblocks, 64 bh).
// Pass 1 stages each 8KB K-chunk into LDS ONCE per block via async-LDS copies
// (double-buffered, counted s_wait_asynccnt) instead of 4x redundant global loads.
// Causal skip: chunks beyond the block's last row are skipped in pass 1; in pass 2,
// per-wave fully-masked chunks store exact zeros with no reload/WMMA.
__global__ __launch_bounds__(128) void attn_fused(
    const unsigned short* __restrict__ qh, const unsigned short* __restrict__ kh,
    const unsigned short* __restrict__ vt, float* __restrict__ attn,
    unsigned short* __restrict__ concat)
{
  __shared__ alignas(16) unsigned short kbuf[2][64 * 64];   // 2 x 8KB K-chunks
  __shared__ float m_lds[64], l_lds[64];

  const int tid = threadIdx.x, lane = tid & 31, wid = tid >> 5;
  const int lane15 = lane & 15, kg = lane >> 4;
  const int bh = blockIdx.y;                    // b*16 + h
  const int q0 = blockIdx.x * 64 + wid * 16;
  const int imax = q0 + 15;                     // last query row of this wave

  // Q fragment: 16 rows x DEPTH=64 (two K=32 chunks), A-operand layout
  const unsigned short* qrow = qh + ((size_t)bh * 2048 + q0 + lane15) * 64;
  const v16bf qa0 = frag_cast(*(const uint4*)(qrow + 8 * kg),
                              *(const uint4*)(qrow + 16 + 8 * kg));
  const v16bf qa1 = frag_cast(*(const uint4*)(qrow + 32 + 8 * kg),
                              *(const uint4*)(qrow + 48 + 8 * kg));

  float* abase = attn + (size_t)bh * 2048 * 2048;
  const v8f vzero = {0.f,0.f,0.f,0.f,0.f,0.f,0.f,0.f};

  float m_run[8], l_run[8];
#pragma unroll
  for (int r = 0; r < 8; ++r) { m_run[r] = -3.0e38f; l_run[r] = 0.0f; }

  // async-stage a 64-key x 64-depth bf16 chunk (8KB contiguous) into kbuf[p]
  const unsigned long long khbase =
      (unsigned long long)(uintptr_t)(kh + (size_t)bh * 2048 * 64);
  auto stage = [&](int p, int kc) {
    const unsigned long long g0 = khbase + (unsigned long long)kc * 128ull;
    const unsigned lb = (unsigned)(uintptr_t)(&kbuf[p][0]);
#pragma unroll
    for (int i = 0; i < 4; ++i) {
      const unsigned off = (unsigned)((i * 128 + tid) * 16);
      async_copy_b128(lb + off, g0 + off);
    }
  };

  // ---- pass 1: block-uniform chunk count so staging barriers stay uniform ----
  const int nchunks = blockIdx.x + 1;           // chunks of 64 keys
  stage(0, 0);
  for (int ci = 0; ci < nchunks; ++ci) {
    const int kc = ci * 64;
    const int p  = ci & 1;
    const bool more = (ci + 1) < nchunks;
    if (more) { stage(p ^ 1, kc + 64); wait_async_le4(); }  // prev chunk landed
    else      { wait_async_0(); }
    __syncthreads();                                        // all waves' copies done

    v8f a[4];
#pragma unroll
    for (int t = 0; t < 4; ++t) {
      const unsigned short* kl = &kbuf[p][(t * 16 + lane15) * 64 + 8 * kg];
      v16bf b0 = frag_cast(*(const uint4*)(kl),      *(const uint4*)(kl + 16));
      v16bf b1 = frag_cast(*(const uint4*)(kl + 32), *(const uint4*)(kl + 48));
      a[t] = __builtin_amdgcn_wmma_f32_16x16x32_bf16(
          false, qa0, false, b0, (short)0, vzero, false, false);
      a[t] = __builtin_amdgcn_wmma_f32_16x16x32_bf16(
          false, qa1, false, b1, (short)0, a[t], false, false);
    }

    float xa[4][8];
#pragma unroll
    for (int t = 0; t < 4; ++t) {
      const int j = kc + t * 16 + lane15;
#pragma unroll
      for (int r = 0; r < 8; ++r) {
        const int i = q0 + 8 * kg + r;
        float g = gelu_exact(a[t][r] * 0.125f);   // / sqrt(64), exact gelu
        if (j > i) g += MASK_NEG;                 // causal
        abase[(size_t)i * 2048 + j] = g;
        xa[t][r] = g;
      }
    }
#pragma unroll
    for (int r = 0; r < 8; ++r) {
      float cm = fmaxf(fmaxf(xa[0][r], xa[1][r]), fmaxf(xa[2][r], xa[3][r]));
#pragma unroll
      for (int m = 1; m < 16; m <<= 1) cm = fmaxf(cm, __shfl_xor(cm, m, 32));
      const float mn = fmaxf(m_run[r], cm);
      float se = __expf(xa[0][r] - mn) + __expf(xa[1][r] - mn) +
                 __expf(xa[2][r] - mn) + __expf(xa[3][r] - mn);
#pragma unroll
      for (int m = 1; m < 16; m <<= 1) se += __shfl_xor(se, m, 32);
      l_run[r] = l_run[r] * __expf(m_run[r] - mn) + se;
      m_run[r] = mn;
    }
    __syncthreads();   // buffer p may be re-staged next iteration
  }

  // exchange per-row (m, l) through LDS for pass-2 lane layout
  if (lane15 == 0) {
#pragma unroll
    for (int r = 0; r < 8; ++r) {
      m_lds[wid * 16 + 8 * kg + r] = m_run[r];
      l_lds[wid * 16 + 8 * kg + r] = l_run[r];
    }
  }
  __syncthreads();
  const float mrow = m_lds[wid * 16 + lane15];
  const float linv = 1.0f / l_lds[wid * 16 + lane15];

  // ---- pass 2: normalize attn in place, ctx = attn * V via WMMA ----
  v8f cacc[4];
#pragma unroll
  for (int t = 0; t < 4; ++t) cacc[t] = vzero;

  float* arow = abase + (size_t)(q0 + lane15) * 2048;   // this lane's query row
  const float4 fz = make_float4(0.f, 0.f, 0.f, 0.f);

  for (int kc = 0; kc < 2048; kc += 32) {
    if (kc > imax) {  // fully masked: attn rows are exactly zero, no WMMA needed
      *(float4*)(arow + kc + 8 * kg)      = fz;
      *(float4*)(arow + kc + 8 * kg + 4)  = fz;
      *(float4*)(arow + kc + 16 + 8 * kg) = fz;
      *(float4*)(arow + kc + 20 + 8 * kg) = fz;
      continue;
    }
    float xv[16];
    *(float4*)&xv[0]  = *(const float4*)(arow + kc + 8 * kg);
    *(float4*)&xv[4]  = *(const float4*)(arow + kc + 8 * kg + 4);
    *(float4*)&xv[8]  = *(const float4*)(arow + kc + 16 + 8 * kg);
    *(float4*)&xv[12] = *(const float4*)(arow + kc + 20 + 8 * kg);
    // batch V B-fragment loads before the exp/pack so they overlap the VALU work
    uint4 vf[4][2];
#pragma unroll
    for (int t = 0; t < 4; ++t) {
      const unsigned short* vrow = vt + ((size_t)bh * 64 + t * 16 + lane15) * 2048
                                      + kc + 8 * kg;
      vf[t][0] = *(const uint4*)vrow; vf[t][1] = *(const uint4*)(vrow + 16);
    }
#pragma unroll
    for (int e = 0; e < 16; ++e) xv[e] = __expf(xv[e] - mrow) * linv;
    *(float4*)(arow + kc + 8 * kg)      = *(const float4*)&xv[0];
    *(float4*)(arow + kc + 8 * kg + 4)  = *(const float4*)&xv[4];
    *(float4*)(arow + kc + 16 + 8 * kg) = *(const float4*)&xv[8];
    *(float4*)(arow + kc + 20 + 8 * kg) = *(const float4*)&xv[12];

    unsigned int pk[8];
#pragma unroll
    for (int e = 0; e < 8; ++e) pk[e] = pkbf(xv[2 * e], xv[2 * e + 1]);
    v16bf pa = frag_cast(make_uint4(pk[0], pk[1], pk[2], pk[3]),
                         make_uint4(pk[4], pk[5], pk[6], pk[7]));
#pragma unroll
    for (int t = 0; t < 4; ++t) {
      cacc[t] = __builtin_amdgcn_wmma_f32_16x16x32_bf16(
          false, pa, false, frag_cast(vf[t][0], vf[t][1]), (short)0, cacc[t], false, false);
    }
  }

  // concat[b][s][h*64+d] = ctx, bf16 for final projection
  const int b = bh >> 4, h = bh & 15;
#pragma unroll
  for (int t = 0; t < 4; ++t) {
#pragma unroll
    for (int r = 0; r < 8; ++r) {
      const int i = q0 + 8 * kg + r;
      const int col = h * 64 + t * 16 + lane15;
      concat[((size_t)b * 2048 + i) * 1024 + col] = f2bf(cacc[t][r]);
    }
  }
}

extern "C" void kernel_launch(void* const* d_in, const int* in_sizes, int n_in,
                              void* d_out, int out_size, void* d_ws, size_t ws_size,
                              hipStream_t stream) {
  (void)in_sizes; (void)n_in; (void)out_size; (void)ws_size;
  const float* q  = (const float*)d_in[0];
  const float* k  = (const float*)d_in[1];
  const float* v  = (const float*)d_in[2];
  // d_in[3] = mask (causal triu, recomputed analytically in-kernel)
  const float* Wq = (const float*)d_in[4];
  const float* bq = (const float*)d_in[5];
  const float* Wk = (const float*)d_in[6];
  const float* bk = (const float*)d_in[7];
  const float* Wv = (const float*)d_in[8];
  const float* bv = (const float*)d_in[9];
  const float* Wo = (const float*)d_in[10];
  const float* bo = (const float*)d_in[11];

  char* ws = (char*)d_ws;
  unsigned short* qh = (unsigned short*)(ws);
  unsigned short* kh = (unsigned short*)(ws + (size_t)16 * 1024 * 1024);
  unsigned short* vt = (unsigned short*)(ws + (size_t)32 * 1024 * 1024);
  unsigned short* cc = (unsigned short*)(ws + (size_t)48 * 1024 * 1024);

  float* out  = (float*)d_out;
  float* attn = out + (size_t)8192 * 1024;

  dim3 gp(128, 8);
  gemm_proj<float, 0><<<gp, 256, 0, stream>>>(q, Wq, bq, qh);
  gemm_proj<float, 0><<<gp, 256, 0, stream>>>(k, Wk, bk, kh);
  gemm_proj<float, 1><<<gp, 256, 0, stream>>>(v, Wv, bv, vt);
  attn_fused<<<dim3(32, 64), 128, 0, stream>>>(qh, kh, vt, attn, cc);
  gemm_proj<unsigned short, 2><<<gp, 256, 0, stream>>>(cc, Wo, bo, out);
}
